// CausalAttention_33337536152252
// MI455X (gfx1250) — matrix operand order repaired and checked by reading, not compiled, not run on hardware
//
#include <hip/hip_runtime.h>
#include <stdint.h>
#include <math.h>

typedef __bf16 bf16;
typedef __attribute__((ext_vector_type(16))) __bf16 bf16x16;
typedef __attribute__((ext_vector_type(8)))  float  f32x8;

union Frag { bf16x16 v; uint32_t u[8]; };

#define DIM_    1024
#define HEADS_  8
#define DHEAD_  128
#define B_      2
#define N_      2048
#define ROWS_   (B_*N_)        /* 4096 */
#define QKVC_   3072
#define BH_     (B_*HEADS_)    /* 16 */
#define EPS_    1e-10f

static __device__ __forceinline__ f32x8 f32x8_zero() {
  f32x8 z;
#pragma unroll
  for (int i = 0; i < 8; ++i) z[i] = 0.0f;
  return z;
}

static __device__ __forceinline__ float red_max16(float v) {
#pragma unroll
  for (int m = 1; m < 16; m <<= 1) v = fmaxf(v, __shfl_xor(v, m, 32));
  return v;
}
static __device__ __forceinline__ float red_sum16(float v) {
#pragma unroll
  for (int m = 1; m < 16; m <<= 1) v += __shfl_xor(v, m, 32);
  return v;
}
static __device__ __forceinline__ float red_sum32(float v) {
#pragma unroll
  for (int m = 1; m < 32; m <<= 1) v += __shfl_xor(v, m, 32);
  return v;
}

// A-fragment (16x32 bf16, row-major A with leading dim lda elements).
// Lane L: m = L&15; lanes 0-15 hold K 0-7 / 16-23, lanes 16-31 hold K 8-15 / 24-31.
static __device__ __forceinline__ void load_a_global(Frag& a, const bf16* A, int lda, int lane) {
  int l15 = lane & 15, half = lane >> 4;
  const uint32_t* arow = (const uint32_t*)(A + (size_t)l15 * lda);
  int d0 = half * 4; // dword base: k = half*8
#pragma unroll
  for (int j = 0; j < 4; ++j) a.u[j] = arow[d0 + j];
#pragma unroll
  for (int j = 0; j < 4; ++j) a.u[4 + j] = arow[d0 + 8 + j];
}

// B-fragment (32x16 bf16) sourced from BT (B transposed, row-major, ld = ldbt):
// lane L: n = L&15; lanes 0-15 hold K 0-7 / 16-23, lanes 16-31 hold K 8-15 / 24-31.
static __device__ __forceinline__ void load_b_global(Frag& b, const bf16* BT, int ldbt, int lane) {
  int l15 = lane & 15, half = lane >> 4;
  const uint32_t* brow = (const uint32_t*)(BT + (size_t)l15 * ldbt);
  int d0 = half * 4;
#pragma unroll
  for (int j = 0; j < 4; ++j) b.u[j] = brow[d0 + j];
#pragma unroll
  for (int j = 0; j < 4; ++j) b.u[4 + j] = brow[d0 + 8 + j];
}

static __device__ __forceinline__ f32x8 wmma_bf16(const Frag& a, const Frag& b, f32x8 c) {
  return __builtin_amdgcn_wmma_f32_16x16x32_bf16(false, a.v, false, b.v, (short)0, c, false, false);
}

// Software-pipelined row-tile GEMM: one 16-row A stripe vs NT consecutive
// 16-col B tiles (BT rows at BT0 + t*btStride). 2x-unrolled K-loop with two
// named buffer sets (ping-pong) so the prefetch needs no register copies and
// loads overlap WMMA issue. Requires K % 64 == 0.
template <int NT>
static __device__ __forceinline__ void gemm_rowtile(const bf16* A, int lda,
                                                    const bf16* BT0, size_t btStride, int ldbt,
                                                    int K, f32x8* acc, int lane) {
  Frag a0; Frag b0[NT];
  Frag a1; Frag b1[NT];
  load_a_global(a0, A, lda, lane);
#pragma unroll
  for (int t = 0; t < NT; ++t) load_b_global(b0[t], BT0 + btStride * t, ldbt, lane);
  for (int k0 = 0; k0 < K; k0 += 64) {
    // prefetch k0+32 into buffer set 1 (always in range: K % 64 == 0)
    load_a_global(a1, A + k0 + 32, lda, lane);
#pragma unroll
    for (int t = 0; t < NT; ++t) load_b_global(b1[t], BT0 + btStride * t + k0 + 32, ldbt, lane);
#pragma unroll
    for (int t = 0; t < NT; ++t) acc[t] = wmma_bf16(a0, b0[t], acc[t]);
    // prefetch k0+64 into buffer set 0
    if (k0 + 64 < K) {
      load_a_global(a0, A + k0 + 64, lda, lane);
#pragma unroll
      for (int t = 0; t < NT; ++t) load_b_global(b0[t], BT0 + btStride * t + k0 + 64, ldbt, lane);
    }
#pragma unroll
    for (int t = 0; t < NT; ++t) acc[t] = wmma_bf16(a1, b1[t], acc[t]);
  }
}

// ---------------- Kernel 1: L2 norm of x rows -> bf16 ----------------
__global__ void k_norm(const float* __restrict__ x, const float* __restrict__ gamma,
                       bf16* __restrict__ xn) {
  int row = blockIdx.x, t = threadIdx.x;
  const float* xr = x + (size_t)row * DIM_;
  __shared__ float sred[256];
  float ss = 0.f;
  for (int i = t; i < DIM_; i += 256) { float v = xr[i]; ss += v * v; }
  sred[t] = ss; __syncthreads();
  for (int s = 128; s > 0; s >>= 1) { if (t < s) sred[t] += sred[t + s]; __syncthreads(); }
  float scale = 32.0f / fmaxf(sqrtf(sred[0]), 1e-12f); // sqrt(1024)=32
  bf16* o = xn + (size_t)row * DIM_;
  for (int i = t; i < DIM_; i += 256) o[i] = (bf16)(xr[i] * scale * gamma[i]);
}

// ---------------- Kernel 2: transpose-convert W[R][C] -> WT[C][R] bf16 ----------------
__global__ void k_cvt_T(const float* __restrict__ W, bf16* __restrict__ WT, int R, int C) {
  size_t i = (size_t)blockIdx.x * blockDim.x + threadIdx.x;
  if (i >= (size_t)R * C) return;
  int r = (int)(i / C), c = (int)(i % C);
  WT[(size_t)c * R + r] = (bf16)W[i];
}

// past_kv [bh][dk][dv] -> pkT [bh][dv][dk] bf16
__global__ void k_cvt_pkT(const float* __restrict__ pk, bf16* __restrict__ pkT) {
  size_t i = (size_t)blockIdx.x * blockDim.x + threadIdx.x; // BH_*128*128
  int bh = (int)(i >> 14); int rem = (int)(i & 16383); int dk = rem >> 7; int dv = rem & 127;
  pkT[((size_t)bh * DHEAD_ + dv) * DHEAD_ + dk] = (bf16)pk[i];
}

// ---------------- Kernel 3: qkv = xn @ w_qkv (bf16 WMMA, f32 out) ----------------
__global__ void k_gemm_qkv(const bf16* __restrict__ xn, const bf16* __restrict__ wT,
                           float* __restrict__ qkv) {
  int wave = blockIdx.x * 8 + (threadIdx.x >> 5);
  int lane = threadIdx.x & 31, l15 = lane & 15, half = lane >> 4;
  int rowTile = wave / 48, colQuad = wave % 48;
  int r0 = rowTile * 16, c0 = colQuad * 64;
  f32x8 acc[4];
#pragma unroll
  for (int t = 0; t < 4; ++t) acc[t] = f32x8_zero();
  gemm_rowtile<4>(xn + (size_t)r0 * DIM_, DIM_,
                  wT + (size_t)c0 * DIM_, (size_t)16 * DIM_, DIM_, DIM_, acc, lane);
#pragma unroll
  for (int t = 0; t < 4; ++t)
#pragma unroll
    for (int r = 0; r < 8; ++r)
      qkv[(size_t)(r0 + r + 8 * half) * QKVC_ + c0 + 16 * t + l15] = acc[t][r];
}

// ---------------- Kernel 4: RoPE + elu+1 + past_norm dots ----------------
__global__ void k_rope(const float* __restrict__ qkv, const float* __restrict__ past_norm,
                       bf16* __restrict__ qr, bf16* __restrict__ kr,
                       bf16* __restrict__ qe, bf16* __restrict__ ke, bf16* __restrict__ keT,
                       float* __restrict__ mnorm, float* __restrict__ dnorm) {
  int wid = blockIdx.x * 8 + (threadIdx.x >> 5);   // row over B*H*N
  int lane = threadIdx.x & 31;
  int bh = wid / N_, n = wid % N_;
  int b = bh / HEADS_, h = bh % HEADS_;
  const float* qrow = qkv + (size_t)(b * N_ + n) * QKVC_ + (size_t)h * DHEAD_;
  const float* krow = qrow + (size_t)HEADS_ * DHEAD_;
  int d0 = lane * 4;
  const float qscale = 0.08838834764831845f; // 128^-0.5
  float q0 = qrow[d0] * qscale, q1 = qrow[d0 + 1] * qscale;
  float q2 = qrow[d0 + 2] * qscale, q3 = qrow[d0 + 3] * qscale;
  float k0 = krow[d0], k1 = krow[d0 + 1], k2 = krow[d0 + 2], k3 = krow[d0 + 3];
  float f0 = __powf(10000.f, -(float)d0 / 128.f);
  float f1 = __powf(10000.f, -(float)(d0 + 2) / 128.f);
  float a0 = (float)n * f0, a1 = (float)n * f1;
  float c0 = __cosf(a0), s0 = __sinf(a0), c1 = __cosf(a1), s1 = __sinf(a1);
  float Q0 = q0 * c0 - q1 * s0, Q1 = q1 * c0 + q0 * s0;
  float Q2 = q2 * c1 - q3 * s1, Q3 = q3 * c1 + q2 * s1;
  float K0 = k0 * c0 - k1 * s0, K1 = k1 * c0 + k0 * s0;
  float K2 = k2 * c1 - k3 * s1, K3 = k3 * c1 + k2 * s1;
  size_t ro = (size_t)wid * DHEAD_ + d0;
  qr[ro] = (bf16)Q0; qr[ro + 1] = (bf16)Q1; qr[ro + 2] = (bf16)Q2; qr[ro + 3] = (bf16)Q3;
  kr[ro] = (bf16)K0; kr[ro + 1] = (bf16)K1; kr[ro + 2] = (bf16)K2; kr[ro + 3] = (bf16)K3;
  float E0 = Q0 > 0.f ? Q0 + 1.f : __expf(Q0);
  float E1 = Q1 > 0.f ? Q1 + 1.f : __expf(Q1);
  float E2 = Q2 > 0.f ? Q2 + 1.f : __expf(Q2);
  float E3 = Q3 > 0.f ? Q3 + 1.f : __expf(Q3);
  float F0 = K0 > 0.f ? K0 + 1.f : __expf(K0);
  float F1 = K1 > 0.f ? K1 + 1.f : __expf(K1);
  float F2 = K2 > 0.f ? K2 + 1.f : __expf(K2);
  float F3 = K3 > 0.f ? K3 + 1.f : __expf(K3);
  qe[ro] = (bf16)E0; qe[ro + 1] = (bf16)E1; qe[ro + 2] = (bf16)E2; qe[ro + 3] = (bf16)E3;
  ke[ro] = (bf16)F0; ke[ro + 1] = (bf16)F1; ke[ro + 2] = (bf16)F2; ke[ro + 3] = (bf16)F3;
  keT[((size_t)bh * DHEAD_ + d0 + 0) * N_ + n] = (bf16)F0;
  keT[((size_t)bh * DHEAD_ + d0 + 1) * N_ + n] = (bf16)F1;
  keT[((size_t)bh * DHEAD_ + d0 + 2) * N_ + n] = (bf16)F2;
  keT[((size_t)bh * DHEAD_ + d0 + 3) * N_ + n] = (bf16)F3;
  const float* pn = past_norm + (size_t)bh * DHEAD_ + d0;
  float md = E0 * pn[0] + E1 * pn[1] + E2 * pn[2] + E3 * pn[3];
  float dd = F0 * pn[0] + F1 * pn[1] + F2 * pn[2] + F3 * pn[3];
  md = red_sum32(md); dd = red_sum32(dd);
  if (lane == 0) { mnorm[wid] = md; dnorm[wid] = dd; }
}

// ---------------- Kernel 5: mem GEMMs + delta-V ----------------
__global__ void k_mem(const bf16* __restrict__ qe, const bf16* __restrict__ ke,
                      const bf16* __restrict__ pkT, const float* __restrict__ mnorm,
                      const float* __restrict__ dnorm, const float* __restrict__ qkv,
                      float* __restrict__ memo, bf16* __restrict__ vT) {
  int wave = blockIdx.x * 8 + (threadIdx.x >> 5);
  int lane = threadIdx.x & 31, l15 = lane & 15, half = lane >> 4;
  int bh = wave >> 7;
  int n0 = (wave & 127) << 4;
  int b = bh >> 3, h = bh & 7;
  const bf16* BT = pkT + (size_t)bh * DHEAD_ * DHEAD_;
  { // mem_out = qe @ past_kv, normalized
    f32x8 acc[8];
#pragma unroll
    for (int t = 0; t < 8; ++t) acc[t] = f32x8_zero();
    gemm_rowtile<8>(qe + ((size_t)bh * N_ + n0) * DHEAD_, DHEAD_,
                    BT, (size_t)16 * DHEAD_, DHEAD_, DHEAD_, acc, lane);
#pragma unroll
    for (int r = 0; r < 8; ++r) {
      int n = n0 + r + 8 * half;
      float inv = 1.f / fmaxf(mnorm[(size_t)bh * N_ + n], EPS_);
#pragma unroll
      for (int t = 0; t < 8; ++t)
        memo[((size_t)bh * N_ + n) * DHEAD_ + t * 16 + l15] = acc[t][r] * inv;
    }
  }
  { // delta_v = ke @ past_kv ; v' = v - delta_v/clip(dnorm); store v'^T bf16
    f32x8 acc[8];
#pragma unroll
    for (int t = 0; t < 8; ++t) acc[t] = f32x8_zero();
    gemm_rowtile<8>(ke + ((size_t)bh * N_ + n0) * DHEAD_, DHEAD_,
                    BT, (size_t)16 * DHEAD_, DHEAD_, DHEAD_, acc, lane);
#pragma unroll
    for (int r = 0; r < 8; ++r) {
      int n = n0 + r + 8 * half;
      float inv = 1.f / fmaxf(dnorm[(size_t)bh * N_ + n], EPS_);
#pragma unroll
      for (int t = 0; t < 8; ++t) {
        int d = t * 16 + l15;
        float v = qkv[(size_t)(b * N_ + n) * QKVC_ + (size_t)(2 * HEADS_ + h) * DHEAD_ + d];
        vT[((size_t)bh * DHEAD_ + d) * N_ + n] = (bf16)(v - acc[t][r] * inv);
      }
    }
  }
}

// ---------------- Kernel 6: causal flash attention + gate fuse ----------------
__global__ void k_attn(const bf16* __restrict__ qr, const bf16* __restrict__ kr,
                       const bf16* __restrict__ vT, const float* __restrict__ memo,
                       const float* __restrict__ head_gates, bf16* __restrict__ comb) {
  __shared__ uint32_t pbuf[8 * 256]; // per wave: 16x32 bf16 P tile = 256 dwords
  int wv = threadIdx.x >> 5, lane = threadIdx.x & 31, l15 = lane & 15, half = lane >> 4;
  int wave = blockIdx.x * 8 + wv;
  int bh = wave >> 7;
  int q0 = (wave & 127) << 4;
  int b = bh >> 3, h = bh & 7;
  const bf16* Q = qr + ((size_t)bh * N_ + q0) * DHEAD_;
  uint32_t* P = pbuf + wv * 256;
  f32x8 O[8];
#pragma unroll
  for (int t = 0; t < 8; ++t) O[t] = f32x8_zero();
  float mrow[8], lrow[8];
#pragma unroll
  for (int r = 0; r < 8; ++r) { mrow[r] = -1e30f; lrow[r] = 0.f; }

  for (int jb = 0; jb < q0 + 16; jb += 32) {
    // S tile pair (pipelined over contraction dim d)
    f32x8 s[2]; s[0] = f32x8_zero(); s[1] = f32x8_zero();
    gemm_rowtile<2>(Q, DHEAD_,
                    kr + ((size_t)bh * N_ + jb) * DHEAD_, (size_t)16 * DHEAD_, DHEAD_,
                    DHEAD_, s, lane);
#pragma unroll
    for (int r = 0; r < 8; ++r) {
      int i = q0 + r + 8 * half;
      float v0 = (jb + l15 > i) ? -1e30f : s[0][r];
      float v1 = (jb + 16 + l15 > i) ? -1e30f : s[1][r];
      float rm = red_max16(fmaxf(v0, v1));
      float mnew = fmaxf(mrow[r], rm);
      float alpha = __expf(mrow[r] - mnew);
      float p0 = __expf(v0 - mnew), p1 = __expf(v1 - mnew);
      lrow[r] = lrow[r] * alpha + red_sum16(p0 + p1);
      mrow[r] = mnew;
#pragma unroll
      for (int t = 0; t < 8; ++t) O[t][r] *= alpha;
      int m = r + 8 * half;
      bf16* prow = (bf16*)P + m * 32;
      prow[l15] = (bf16)p0;
      prow[l15 + 16] = (bf16)p1;
    }
    __builtin_amdgcn_wave_barrier();
    asm volatile("s_wait_dscnt 0" ::: "memory");
    Frag ap;
    {
      const uint32_t* prow = P + l15 * 16;
      int d0 = half * 4;
#pragma unroll
      for (int j = 0; j < 4; ++j) ap.u[j] = prow[d0 + j];
#pragma unroll
      for (int j = 0; j < 4; ++j) ap.u[4 + j] = prow[d0 + 8 + j];
    }
    __builtin_amdgcn_wave_barrier();
    // P@V: batch all 8 V^T fragments (one load clause), then 8 WMMAs.
    Frag bv[8];
#pragma unroll
    for (int t = 0; t < 8; ++t)
      load_b_global(bv[t], vT + ((size_t)bh * DHEAD_ + 16 * t) * N_ + jb, N_, lane);
#pragma unroll
    for (int t = 0; t < 8; ++t) O[t] = wmma_bf16(ap, bv[t], O[t]);
  }
  float g = 1.f / (1.f + __expf(-head_gates[h]));
#pragma unroll
  for (int r = 0; r < 8; ++r) {
    int i = q0 + r + 8 * half;
    float invl = 1.0f / lrow[r];
#pragma unroll
    for (int t = 0; t < 8; ++t) {
      int d = 16 * t + l15;
      float att = O[t][r] * invl;
      float mo = memo[((size_t)bh * N_ + i) * DHEAD_ + d];
      comb[(size_t)(b * N_ + i) * DIM_ + h * DHEAD_ + d] = (bf16)(att * g + mo * (1.f - g));
    }
  }
}

// ---------------- Kernel 7: new_kv = ke^T @ v' + past_kv ----------------
__global__ void k_newkv(const bf16* __restrict__ keT, const bf16* __restrict__ vT,
                        const float* __restrict__ past_kv, float* __restrict__ out_kv) {
  int wv = threadIdx.x >> 5, lane = threadIdx.x & 31, l15 = lane & 15, half = lane >> 4;
  int bh = blockIdx.x;
  for (int tt = 0; tt < 8; ++tt) {
    int tile = wv * 8 + tt;
    int m0 = (tile >> 3) * 16, c0 = (tile & 7) * 16;
    f32x8 acc[1]; acc[0] = f32x8_zero();
    gemm_rowtile<1>(keT + ((size_t)bh * DHEAD_ + m0) * N_, N_,
                    vT + ((size_t)bh * DHEAD_ + c0) * N_, 0, N_, N_, acc, lane);
#pragma unroll
    for (int r = 0; r < 8; ++r) {
      size_t idx = ((size_t)bh * DHEAD_ + m0 + r + 8 * half) * DHEAD_ + c0 + l15;
      out_kv[idx] = acc[0][r] + past_kv[idx];
    }
  }
}

// ---------------- Kernel 8: new_norm = sum_n ke + past_norm ----------------
__global__ void k_newnorm(const bf16* __restrict__ keT, const float* __restrict__ past_norm,
                          float* __restrict__ out_norm) {
  int t = blockIdx.x * blockDim.x + threadIdx.x; // BH_*DHEAD_
  const bf16* row = keT + (size_t)t * N_;
  float s = 0.f;
  for (int n = 0; n < N_; ++n) s += (float)row[n];
  out_norm[t] = s + past_norm[t];
}

// ---------------- Kernel 9: out = comb @ w_out ----------------
__global__ void k_gemm_out(const bf16* __restrict__ comb, const bf16* __restrict__ wT,
                           float* __restrict__ out) {
  int wave = blockIdx.x * 8 + (threadIdx.x >> 5);
  int lane = threadIdx.x & 31, l15 = lane & 15, half = lane >> 4;
  int rowTile = wave >> 4, colQuad = wave & 15;
  int r0 = rowTile * 16, c0 = colQuad * 64;
  f32x8 acc[4];
#pragma unroll
  for (int t = 0; t < 4; ++t) acc[t] = f32x8_zero();
  gemm_rowtile<4>(comb + (size_t)r0 * DIM_, DIM_,
                  wT + (size_t)c0 * DIM_, (size_t)16 * DIM_, DIM_, DIM_, acc, lane);
#pragma unroll
  for (int t = 0; t < 4; ++t)
#pragma unroll
    for (int r = 0; r < 8; ++r)
      out[(size_t)(r0 + r + 8 * half) * DIM_ + c0 + 16 * t + l15] = acc[t][r];
}

extern "C" void kernel_launch(void* const* d_in, const int* in_sizes, int n_in,
                              void* d_out, int out_size, void* d_ws, size_t ws_size,
                              hipStream_t stream) {
  const float* x         = (const float*)d_in[0];
  const float* gamma     = (const float*)d_in[1];
  const float* w_qkv     = (const float*)d_in[2];
  const float* w_out     = (const float*)d_in[3];
  const float* gates     = (const float*)d_in[4];
  const float* past_kv   = (const float*)d_in[5];
  const float* past_norm = (const float*)d_in[6];

  float* out      = (float*)d_out;
  float* out_kv   = out + (size_t)ROWS_ * DIM_;
  float* out_norm = out_kv + (size_t)BH_ * DHEAD_ * DHEAD_;

  char* w = (char*)d_ws; size_t off = 0;
  auto alloc = [&](size_t bytes) { void* p = w + off; off += (bytes + 255) & ~(size_t)255; return p; };
  bf16* xn    = (bf16*)alloc((size_t)ROWS_ * DIM_ * 2);
  bf16* wqkvT = (bf16*)alloc((size_t)QKVC_ * DIM_ * 2);
  bf16* woutT = (bf16*)alloc((size_t)DIM_ * DIM_ * 2);
  bf16* pkT   = (bf16*)alloc((size_t)BH_ * DHEAD_ * DHEAD_ * 2);
  float* qkv  = (float*)alloc((size_t)ROWS_ * QKVC_ * 4);
  bf16* qr    = (bf16*)alloc((size_t)BH_ * N_ * DHEAD_ * 2);
  bf16* kr    = (bf16*)alloc((size_t)BH_ * N_ * DHEAD_ * 2);
  bf16* qe    = (bf16*)alloc((size_t)BH_ * N_ * DHEAD_ * 2);
  bf16* ke    = (bf16*)alloc((size_t)BH_ * N_ * DHEAD_ * 2);
  bf16* keT   = (bf16*)alloc((size_t)BH_ * DHEAD_ * N_ * 2);
  bf16* vT    = (bf16*)alloc((size_t)BH_ * DHEAD_ * N_ * 2);
  float* memo = (float*)alloc((size_t)BH_ * N_ * DHEAD_ * 4);
  float* mnorm = (float*)alloc((size_t)BH_ * N_ * 4);
  float* dnorm = (float*)alloc((size_t)BH_ * N_ * 4);
  bf16* comb  = (bf16*)alloc((size_t)ROWS_ * DIM_ * 2);

  k_norm<<<ROWS_, 256, 0, stream>>>(x, gamma, xn);
  k_cvt_T<<<(DIM_ * QKVC_ + 255) / 256, 256, 0, stream>>>(w_qkv, wqkvT, DIM_, QKVC_);
  k_cvt_T<<<(DIM_ * DIM_ + 255) / 256, 256, 0, stream>>>(w_out, woutT, DIM_, DIM_);
  k_cvt_pkT<<<(BH_ * DHEAD_ * DHEAD_) / 256, 256, 0, stream>>>(past_kv, pkT);
  k_gemm_qkv<<<(ROWS_ / 16) * (QKVC_ / 64) / 8, 256, 0, stream>>>(xn, wqkvT, qkv);
  k_rope<<<(BH_ * N_) / 8, 256, 0, stream>>>(qkv, past_norm, qr, kr, qe, ke, keT, mnorm, dnorm);
  k_mem<<<(BH_ * (N_ / 16)) / 8, 256, 0, stream>>>(qe, ke, pkT, mnorm, dnorm, qkv, memo, vT);
  k_attn<<<(BH_ * (N_ / 16)) / 8, 256, 0, stream>>>(qr, kr, vT, memo, gates, comb);
  k_newkv<<<BH_, 256, 0, stream>>>(keT, vT, past_kv, out_kv);
  k_newnorm<<<(BH_ * DHEAD_) / 256, 256, 0, stream>>>(keT, past_norm, out_norm);
  k_gemm_out<<<(ROWS_ / 16) * (DIM_ / 64) / 8, 256, 0, stream>>>(comb, woutT, out);
}
